// ModConv_87359634801320
// MI455X (gfx1250) — matrix operand order repaired and checked by reading, not compiled, not run on hardware
//
#include <hip/hip_runtime.h>
#include <hip/hip_bf16.h>
#include <math.h>

// ---------------------------------------------------------------------------
// Problem constants (from reference)
// ---------------------------------------------------------------------------
#define B_     16
#define CIN_   256
#define COUT_  256
#define HW_    4096
#define WDIM_  64
#define COND_  512

// halo tile geometry for the B (im2col) LDS tile: 4 rows x 68 cols (padded)
#define HALO_X 68
#define HALO_ROWS 272            // 4 * 68
#define HALO_ELEMS (HALO_ROWS * 32)   // 8704 bf16 = 17KB

typedef __attribute__((ext_vector_type(16))) __bf16 v16bf;
typedef __attribute__((ext_vector_type(8)))  __bf16 v8bf;
typedef __attribute__((ext_vector_type(8)))  float  v8f;

// ---------------------------------------------------------------------------
// Kernel 1: cond MLP -> scale[B, CIN]   (tiny, VALU f32)
// ---------------------------------------------------------------------------
__global__ __launch_bounds__(256)
void mlp_scale_kernel(const float* __restrict__ cond,
                      const float* __restrict__ w0, const float* __restrict__ b0,
                      const float* __restrict__ w1, const float* __restrict__ b1,
                      const float* __restrict__ wout, const float* __restrict__ bout,
                      float* __restrict__ scale) {
  __shared__ float sc[COND_];
  __shared__ float h0[COND_];
  __shared__ float h1[COND_];
  const int b = blockIdx.x;
  const int t = threadIdx.x;

  sc[t]       = cond[b * COND_ + t];
  sc[t + 256] = cond[b * COND_ + t + 256];
  __syncthreads();

  #pragma unroll
  for (int rep = 0; rep < 2; ++rep) {
    const int o = t + rep * 256;
    float acc = b0[o];
    const float* wr = w0 + (size_t)o * COND_;
    for (int k = 0; k < COND_; ++k) acc += sc[k] * wr[k];
    h0[o] = acc / (1.0f + __expf(-acc));          // SiLU
  }
  __syncthreads();

  #pragma unroll
  for (int rep = 0; rep < 2; ++rep) {
    const int o = t + rep * 256;
    float acc = b1[o];
    const float* wr = w1 + (size_t)o * COND_;
    for (int k = 0; k < COND_; ++k) acc += h0[k] * wr[k];
    h1[o] = acc / (1.0f + __expf(-acc));          // SiLU
  }
  __syncthreads();

  {
    float acc = bout[t] + 1.0f;
    const float* wr = wout + (size_t)t * COND_;
    for (int k = 0; k < COND_; ++k) acc += h1[k] * wr[k];
    scale[b * CIN_ + t] = acc;
  }
}

// ---------------------------------------------------------------------------
// Kernel 2: modulate + demodulate + pack to bf16 GEMM layout
//   packed[b][tap][o][i] = bf16( conv_w[o,i,tap] * scale[b,i] * demod[b,o] )
// ---------------------------------------------------------------------------
__global__ __launch_bounds__(256)
void modpack_kernel(const float* __restrict__ conv_w,
                    const float* __restrict__ scale,
                    __bf16* __restrict__ packed) {
  const int bo = blockIdx.x;          // b*256 + o
  const int b = bo >> 8;
  const int o = bo & 255;
  const int i = threadIdx.x;

  const float s = scale[b * CIN_ + i];
  float wmod[9];
  float ss = 0.0f;
  const float* wr = conv_w + ((size_t)o * CIN_ + i) * 9;
  #pragma unroll
  for (int t9 = 0; t9 < 9; ++t9) {
    const float v = wr[t9] * s;
    wmod[t9] = v;
    ss += v * v;
  }

  __shared__ float red[256];
  red[i] = ss;
  __syncthreads();
  #pragma unroll
  for (int off = 128; off > 0; off >>= 1) {
    if (i < off) red[i] += red[i + off];
    __syncthreads();
  }
  const float demod = rsqrtf(red[0] + 1e-8f);

  #pragma unroll
  for (int t9 = 0; t9 < 9; ++t9) {
    packed[(((size_t)(b * 9 + t9)) * COUT_ + o) * CIN_ + i] =
        (__bf16)(wmod[t9] * demod);
  }
}

// ---------------------------------------------------------------------------
// Kernel 3: implicit-GEMM conv via v_wmma_f32_16x16x32_bf16
//   Per block: one sample b, 128 output channels x 128 pixels (2 image rows).
//   Channel-block loop (K=32): stage X halo tile ONCE, then run all 9 taps
//   from LDS with shifted fragment rows. Weights staged 3 taps (one kernel
//   row) at a time with vectorized 32B copies.
//   8 waves; each wave: 4x2 tiles of 16x16 (64 o x 32 p).
// ---------------------------------------------------------------------------
__global__ __launch_bounds__(256)
void wmma_conv_kernel(const float* __restrict__ X,
                      const __bf16* __restrict__ packed,
                      float* __restrict__ out) {
  // A: 3 taps x [128 o][32 k]   (24KB)
  __shared__ __attribute__((aligned(64))) __bf16 ldsA[3 * 128 * 32];
  // B: halo [(yh*68+xh)][32 k]  (17KB), zero-padded borders baked in
  __shared__ __attribute__((aligned(64))) __bf16 ldsB[HALO_ELEMS];

  const int tid  = threadIdx.x;
  const int lane = tid & 31;
  const int wid  = tid >> 5;

  const int bid   = blockIdx.x;        // 16 * 2 * 32 = 1024 blocks
  const int b     = bid >> 6;
  const int rr_   = bid & 63;
  const int oTile = (rr_ >> 5) * 128;  // 0 or 128
  const int y0    = (rr_ & 31) * 2;    // output pixel-row pair

  const int waveO    = (wid >> 2) * 64;   // 0 or 64
  const int waveP    = (wid & 3) * 32;    // 0..96
  const int sel      = lane >> 4;         // lane half -> K sub-range
  const int halfLane = lane & 15;

  // per-thread constants for vectorized A staging
  const int oo   = tid >> 1;   // 0..127
  const int half = tid & 1;    // which 16-element half of the K=32 row

  // per-wave fragment LDS row bases (pixel index within the 2x64 tile)
  int pRow[2];
  #pragma unroll
  for (int fb = 0; fb < 2; ++fb) {
    const int p = waveP + fb * 16 + halfLane;
    pRow[fb] = (p >> 6) * HALO_X + (p & 63);   // halo row for tap (0,0)
  }

  v8f acc[4][2];
  #pragma unroll
  for (int fa = 0; fa < 4; ++fa)
    #pragma unroll
    for (int fb = 0; fb < 2; ++fb)
      #pragma unroll
      for (int q = 0; q < 8; ++q) acc[fa][fb][q] = 0.0f;

  const __bf16* wbase = packed + (size_t)b * 9 * COUT_ * CIN_ + (size_t)oTile * CIN_;

  for (int i0 = 0; i0 < CIN_; i0 += 32) {
    // ---- stage B halo tile once per channel block ----
    __syncthreads();   // previous iteration's ldsB readers done
    {
      const float* Xb = X + ((size_t)b * CIN_ + i0) * HW_;
      #pragma unroll
      for (int j = 0; j < 34; ++j) {          // 34*256 == 8704 elements
        const int e   = j * 256 + tid;
        const int ii  = e / HALO_ROWS;        // channel within block (0..31)
        const int ph  = e - ii * HALO_ROWS;   // halo pixel (coalesced in x)
        const int yh  = ph / HALO_X;
        const int xh  = ph - yh * HALO_X;
        const int gy  = y0 + yh - 1;
        const int gx  = xh - 1;
        float v = 0.0f;
        if ((unsigned)gy < 64u && (unsigned)gx < 64u)
          v = Xb[(size_t)ii * HW_ + gy * WDIM_ + gx];
        ldsB[ph * 32 + ii] = (__bf16)v;
      }
    }

    // ---- loop kernel rows: stage 3 taps of weights, run 3x8 WMMA ----
    for (int kh = 0; kh < 3; ++kh) {
      __syncthreads();   // previous ldsA readers done / ldsB stage visible
      #pragma unroll
      for (int kw = 0; kw < 3; ++kw) {
        const int tap = kh * 3 + kw;
        const __bf16* src =
            wbase + ((size_t)tap * COUT_ + oo) * CIN_ + i0 + half * 16;
        *(v16bf*)(ldsA + (kw * 128 + oo) * 32 + half * 16) =
            *(const v16bf*)src;
        if (i0 + 32 < CIN_)
          __builtin_prefetch(src + 32, 0, 1);   // global_prefetch next K-block
      }
      __syncthreads();   // ldsA stage visible

      #pragma unroll
      for (int kw = 0; kw < 3; ++kw) {
        // B fragments: lane = pixel, K run of 16 per lane half
        v16bf bfrag[2];
        #pragma unroll
        for (int fb = 0; fb < 2; ++fb) {
          const int ph = pRow[fb] + kh * HALO_X + kw;   // shifted halo row
          bfrag[fb] = *(const v16bf*)(ldsB + ph * 32 + sel * 16);
        }
        // A fragments + WMMA
        #pragma unroll
        for (int fa = 0; fa < 4; ++fa) {
          const int orow = waveO + fa * 16 + halfLane;
          const __bf16* ar = ldsA + (kw * 128 + orow) * 32 + sel * 16;
          const v8bf lo = *(const v8bf*)(ar);       // K 0..7  / 16..23
          const v8bf hi = *(const v8bf*)(ar + 8);   // K 8..15 / 24..31
          const v16bf afrag = __builtin_shufflevector(
              lo, hi, 0, 1, 2, 3, 4, 5, 6, 7, 8, 9, 10, 11, 12, 13, 14, 15);
          #pragma unroll
          for (int fb = 0; fb < 2; ++fb) {
            acc[fa][fb] = __builtin_amdgcn_wmma_f32_16x16x32_bf16(
                false, afrag, false, bfrag[fb], (short)0, acc[fa][fb],
                false, false);
          }
        }
      }
    }
  }

  // ---- epilogue: C/D layout -> out[b][o][p] (f32) ----
  const int pBase = y0 * WDIM_ + waveP;
  #pragma unroll
  for (int fa = 0; fa < 4; ++fa) {
    #pragma unroll
    for (int fb = 0; fb < 2; ++fb) {
      #pragma unroll
      for (int r = 0; r < 8; ++r) {
        const int o = oTile + waveO + fa * 16 + r + sel * 8;
        const int p = pBase + fb * 16 + halfLane;
        out[((size_t)(b * COUT_ + o)) * HW_ + p] = acc[fa][fb][r];
      }
    }
  }
}

// ---------------------------------------------------------------------------
// Launch
// ---------------------------------------------------------------------------
extern "C" void kernel_launch(void* const* d_in, const int* in_sizes, int n_in,
                              void* d_out, int out_size, void* d_ws, size_t ws_size,
                              hipStream_t stream) {
  const float* X    = (const float*)d_in[0];
  const float* cond = (const float*)d_in[1];
  const float* cw   = (const float*)d_in[2];
  const float* w0   = (const float*)d_in[3];
  const float* b0   = (const float*)d_in[4];
  const float* w1   = (const float*)d_in[5];
  const float* b1   = (const float*)d_in[6];
  const float* wout = (const float*)d_in[7];
  const float* bout = (const float*)d_in[8];
  float* out = (float*)d_out;

  // workspace: scale[16*256] f32 (16KB), then packed bf16 weights (~18.9MB)
  float*  scale  = (float*)d_ws;
  __bf16* packed = (__bf16*)((char*)d_ws + 16384);

  mlp_scale_kernel<<<B_, 256, 0, stream>>>(cond, w0, b0, w1, b1, wout, bout, scale);
  modpack_kernel<<<B_ * COUT_, 256, 0, stream>>>(cw, scale, packed);
  wmma_conv_kernel<<<B_ * 2 * 32, 256, 0, stream>>>(X, packed, out);
}
